// Model_23278722744811
// MI455X (gfx1250) — matrix operand order repaired and checked by reading, not compiled, not run on hardware
//
#include <hip/hip_runtime.h>
#include <hip/hip_bf16.h>

typedef __attribute__((ext_vector_type(16))) _Float16 v16h;
typedef __attribute__((ext_vector_type(8)))  float    v8f;
typedef __attribute__((ext_vector_type(4)))  float    v4f;
typedef __attribute__((ext_vector_type(4)))  unsigned v4u;
typedef __attribute__((ext_vector_type(8)))  int      v8i;
typedef __attribute__((ext_vector_type(4)))  int      v4i;

// ---------------------------------------------------------------------------
// WMMA helper (CDNA5 gfx1250, wave32, V_WMMA_F32_16X16X32_F16)
// ---------------------------------------------------------------------------
__device__ __forceinline__ v8f wmma16(v16h a, v16h b, v8f c) {
    return __builtin_amdgcn_wmma_f32_16x16x32_f16(
        /*neg_a=*/false, a, /*neg_b=*/false, b,
        /*c_mod=*/(short)0, c, /*reuse_a=*/false, /*reuse_b=*/false);
}

// A fragment from f32 global memory (row-major, ld floats): 16x32 f16 tile.
// ISA layout: lane row M=lane&15; halves 0-7 -> K=base+0..7, halves 8-15 ->
// K=base+16..23 with base = kb*32 + (lane<16 ? 0 : 8).  Loads are 4x b128.
__device__ __forceinline__ v16h load_a_frag_f32(const float* __restrict__ X,
                                                int row0, int ld, int kb, int lane) {
    int m     = lane & 15;
    int kbase = kb * 32 + ((lane < 16) ? 0 : 8);
    const v4f* q = (const v4f*)(X + (size_t)(row0 + m) * ld + kbase);
    v4f x0 = q[0], x1 = q[1], x2 = q[4], x3 = q[5];  // p[0..7], p[16..23]
    v16h a;
#pragma unroll
    for (int h = 0; h < 4; ++h) {
        a[h]      = (_Float16)x0[h];
        a[4 + h]  = (_Float16)x1[h];
        a[8 + h]  = (_Float16)x2[h];
        a[12 + h] = (_Float16)x3[h];
    }
    return a;
}

// B fragment from TRANSPOSED f16 weights in LDS: Wt[n][k], ldk halves per row.
// Lane needs col n = nb*16 + lane&15, K = kbase..kbase+7 and kbase+16..kbase+23
// (kbase = kb*32 + (lane<16 ? 0 : 8)) -> 2x ds_load_b128.
__device__ __forceinline__ v16h load_b_frag_t(const _Float16* __restrict__ Wt,
                                              int nb, int kb, int lane, int ldk) {
    typedef __attribute__((ext_vector_type(8))) _Float16 v8h;
    int n     = nb * 16 + (lane & 15);
    int kbase = kb * 32 + ((lane < 16) ? 0 : 8);
    const v8h* p = (const v8h*)(Wt + (size_t)n * ldk + kbase);
    v8h lo = p[0], hi = p[2];
    return __builtin_shufflevector(lo, hi, 0, 1, 2, 3, 4, 5, 6, 7, 8, 9, 10, 11, 12, 13, 14, 15);
}

// C/D tile: VGPR r holds (M = mbase + r, N = lane&15), mbase = 0 / 8.
__device__ __forceinline__ void store_c_bias(float* __restrict__ Y, int row0, int col0,
                                             int ld, int lane, v8f c,
                                             const float* __restrict__ bias) {
    int n     = col0 + (lane & 15);
    int mbase = row0 + ((lane < 16) ? 0 : 8);
    float bv  = bias[n];
#pragma unroll
    for (int r = 0; r < 8; ++r) Y[(size_t)(mbase + r) * ld + n] = c[r] + bv;
}

// ---------------------------------------------------------------------------
// Tensor Data Mover: 2-D f32 tile (dim1 rows x dim0 cols, row stride stride0
// elements) from global -> LDS at byte offset lds_off.  D# per ISA ch.8.
// Issue from ONE wave; follow with s_wait_tensorcnt 0 in that wave.
// clang-23 toolchain builtin: 6 args (g0 v4u, g1 v8i, g2 v4i, g3 v4i, g4 v8i, cpol)
// ---------------------------------------------------------------------------
__device__ __forceinline__ void tdm_load_2d_f32(const float* gsrc, unsigned lds_off,
                                                unsigned dim0, unsigned dim1,
                                                unsigned stride0) {
    unsigned long long ga = (unsigned long long)(size_t)gsrc;
    v4u g0;
    g0[0] = 1u;                                        // count=1, is_restore=0
    g0[1] = lds_off;                                   // lds_addr (bytes)
    g0[2] = (unsigned)(ga & 0xffffffffull);            // global_addr[31:0]
    g0[3] = (unsigned)((ga >> 32) & 0x1ffffffull)      // global_addr[56:32]
            | (2u << 30);                              // type = 2 ("image")
    v8i g1;
    g1[0] = (int)(2u << 16);                           // data_size=2 -> 4 bytes
    g1[1] = (int)((dim0 & 0xffffu) << 16);             // tensor_dim0[15:0]
    g1[2] = (int)(((dim0 >> 16) & 0xffffu) |
                  ((dim1 & 0xffffu) << 16));           // tensor_dim0 hi | dim1 lo
    g1[3] = (int)(((dim1 >> 16) & 0xffffu) |
                  ((dim0 & 0xffffu) << 16));           // dim1 hi | tile_dim0
    g1[4] = (int)(dim1 & 0xffffu);                     // tile_dim1 (tile_dim2=0)
    g1[5] = (int)stride0;                              // tensor_dim0_stride[31:0]
    g1[6] = 0;                                         // stride0 hi | stride1 lo
    g1[7] = 0;
    v4i g2 = {};                                       // 2-D tile: groups 2/3 unused
    v4i g3 = {};
    v8i g4 = {};                                       // extra group (clang-23 form)
    __builtin_amdgcn_tensor_load_to_lds(g0, g1, g2, g3, g4, /*cpol=*/0);
}

// ---------------------------------------------------------------------------
// proj64: Y[nrows x 64] = X[nrows x 64] @ W[64 x 64] + b       (WMMA f16)
// blockDim = 256 (8 waves), each wave owns a 16-row tile -> 128 rows / block.
// W staged via TDM (async tensor load), transposed to f16 in LDS.
// ---------------------------------------------------------------------------
__global__ void proj64(const float* __restrict__ X, const float* __restrict__ W,
                       const float* __restrict__ bias, float* __restrict__ Y,
                       int nrows) {
    __shared__ __align__(16) float    Wstage[64 * 64];  // 16 KB (TDM target)
    __shared__ __align__(32) _Float16 Wt[64 * 64];      //  8 KB, transposed [n][k]

    if (threadIdx.x < 32) {                 // wave 0 issues the tensor DMA
        tdm_load_2d_f32(W, (unsigned)(size_t)Wstage, 64u, 64u, 64u);
        __builtin_amdgcn_s_wait_tensorcnt(0);
    }
    __syncthreads();
    for (int i = threadIdx.x; i < 64 * 64; i += blockDim.x) {
        int k = i >> 6, n = i & 63;
        Wt[n * 64 + k] = (_Float16)Wstage[i];
    }
    __syncthreads();

    int wave = threadIdx.x >> 5;
    int lane = threadIdx.x & 31;
    int row0 = (blockIdx.x * 8 + wave) * 16;
    if (row0 >= nrows) return;

    v8f acc[4] = {};
#pragma unroll
    for (int kb = 0; kb < 2; ++kb) {
        v16h a = load_a_frag_f32(X, row0, 64, kb, lane);
#pragma unroll
        for (int nb = 0; nb < 4; ++nb) {
            v16h b = load_b_frag_t(Wt, nb, kb, lane, 64);
            acc[nb] = wmma16(a, b, acc[nb]);
        }
    }
#pragma unroll
    for (int nb = 0; nb < 4; ++nb)
        store_c_bias(Y, row0, nb * 16, 64, lane, acc[nb], bias);
}

// ---------------------------------------------------------------------------
// Segment softmax attention (3 passes) — L2-resident gather/atomic phase
// ---------------------------------------------------------------------------
__device__ __forceinline__ unsigned f2ord(float f) {
    unsigned u = __float_as_uint(f);
    return (u & 0x80000000u) ? ~u : (u | 0x80000000u);
}
__device__ __forceinline__ float ord2f(unsigned u) {
    return (u & 0x80000000u) ? __uint_as_float(u & 0x7fffffffu)
                             : __uint_as_float(~u);
}

__global__ void init_ms(unsigned* __restrict__ mmax, float* __restrict__ ssum, int n) {
    int i = blockIdx.x * blockDim.x + threadIdx.x;
    if (i < n) { mmax[i] = 0u; ssum[i] = 0.0f; }
}

// wave-per-edge: a[e] = dot(Q[di], K[si] + ea*We) / 8 ; atomicMax segment max
__global__ void attn_logits(const float* __restrict__ Q, const float* __restrict__ K,
                            const int* __restrict__ si, const int* __restrict__ di,
                            const float* __restrict__ ea, const float* __restrict__ We,
                            float* __restrict__ alog, unsigned* __restrict__ mmax, int E) {
    int e    = blockIdx.x * 8 + (threadIdx.x >> 5);
    int lane = threadIdx.x & 31;
    if (e >= E) return;
    int s = si[e], d = di[e];
    float ev = ea[e];
    const float* q = Q + (size_t)d * 64;
    const float* k = K + (size_t)s * 64;
    int c = lane * 2;
    float acc = q[c]     * (k[c]     + ev * We[c]) +
                q[c + 1] * (k[c + 1] + ev * We[c + 1]);
#pragma unroll
    for (int m = 16; m >= 1; m >>= 1) acc += __shfl_xor(acc, m, 32);
    if (lane == 0) {
        float a = acc * 0.125f;  // 1/sqrt(64)
        alog[e] = a;
        atomicMax(&mmax[d], f2ord(a));
    }
}

__global__ void attn_exp(float* __restrict__ alog, const unsigned* __restrict__ mmax,
                         float* __restrict__ ssum, const int* __restrict__ di, int E) {
    int e = blockIdx.x * blockDim.x + threadIdx.x;
    if (e >= E) return;
    int d = di[e];
    float ex = __expf(alog[e] - ord2f(mmax[d]));
    alog[e] = ex;
    atomicAdd(&ssum[d], ex);
}

// 64 threads per edge: OUT[di] += w * (V[si] + ea*We)   (OUT preloaded w/ skip)
__global__ void attn_agg(const float* __restrict__ V, const float* __restrict__ alog,
                         const float* __restrict__ ssum, const float* __restrict__ We,
                         const float* __restrict__ ea, const int* __restrict__ si,
                         const int* __restrict__ di, float* __restrict__ out, int E) {
    int e = blockIdx.x * 4 + (threadIdx.x >> 6);
    int c = threadIdx.x & 63;
    if (e >= E) return;
    int s = si[e], d = di[e];
    float w = alog[e] / ssum[d];
    float v = V[(size_t)s * 64 + c] + ea[e] * We[c];
    atomicAdd(&out[(size_t)d * 64 + c], w * v);
}

__global__ void relu_k(float* __restrict__ x, int n) {
    int i = blockIdx.x * blockDim.x + threadIdx.x;
    if (i < n) x[i] = fmaxf(x[i], 0.0f);
}

// ---------------------------------------------------------------------------
// Decoder: z = relu([z_u[ls] | z_m[ld]] @ W1 + b1); out = z @ W2 + b2
// blockDim = 128 (4 waves), each wave owns 16 labels: 16 WMMAs (K=128, N=64)
// ---------------------------------------------------------------------------
__global__ void decoder(const float* __restrict__ zu, const float* __restrict__ zm,
                        const int* __restrict__ ls, const int* __restrict__ ld,
                        const float* __restrict__ W1, const float* __restrict__ b1,
                        const float* __restrict__ W2, const float* __restrict__ b2,
                        float* __restrict__ out, int L) {
    __shared__ __align__(32) _Float16 W1t[64 * 128];    // 16 KB, transposed [n][k]
    __shared__ float ztile[4][16 * 64];                 // 16 KB (one tile per wave)
    for (int i = threadIdx.x; i < 128 * 64; i += blockDim.x) {
        int k = i >> 6, n = i & 63;                     // W1 is [128][64] row-major
        W1t[n * 128 + k] = (_Float16)W1[i];
    }
    __syncthreads();

    int wave = threadIdx.x >> 5;
    int lane = threadIdx.x & 31;
    int t0   = (blockIdx.x * 4 + wave) * 16;
    if (t0 >= L) return;

    int m  = lane & 15;
    int nu = ls[t0 + m];   // user  node for this fragment row
    int nm = ld[t0 + m];   // movie node for this fragment row
    const float* rowu = zu + (size_t)nu * 64;
    const float* rowm = zm + (size_t)nm * 64;

    v8f acc[4] = {};
#pragma unroll
    for (int kb = 0; kb < 4; ++kb) {
        const float* src = (kb < 2) ? rowu : rowm;
        int cb = (kb & 1) * 32 + ((lane < 16) ? 0 : 8);
        const v4f* q = (const v4f*)(src + cb);
        v4f x0 = q[0], x1 = q[1], x2 = q[4], x3 = q[5];
        v16h a;
#pragma unroll
        for (int h = 0; h < 4; ++h) {
            a[h]      = (_Float16)x0[h];
            a[4 + h]  = (_Float16)x1[h];
            a[8 + h]  = (_Float16)x2[h];
            a[12 + h] = (_Float16)x3[h];
        }
#pragma unroll
        for (int nb = 0; nb < 4; ++nb) {
            v16h b = load_b_frag_t(W1t, nb, kb, lane, 128);
            acc[nb] = wmma16(a, b, acc[nb]);
        }
    }

    // relu(z + b1) -> per-wave LDS tile
    float* zt = ztile[wave];
#pragma unroll
    for (int nb = 0; nb < 4; ++nb) {
        int n     = nb * 16 + (lane & 15);
        int mbase = (lane < 16) ? 0 : 8;
        float bb  = b1[n];
#pragma unroll
        for (int r = 0; r < 8; ++r)
            zt[(mbase + r) * 64 + n] = fmaxf(acc[nb][r] + bb, 0.0f);
    }
    // CDNA5 split-counter wait: make wave's DS stores visible before cross-lane read
    asm volatile("s_wait_dscnt 0" ::: "memory");

    if (lane < 16) {
        float r = b2[0];
#pragma unroll 8
        for (int n = 0; n < 64; ++n) r += zt[lane * 64 + n] * W2[n];
        out[t0 + lane] = r;
    }
}

// ---------------------------------------------------------------------------
// Host driver
// ---------------------------------------------------------------------------
extern "C" void kernel_launch(void* const* d_in, const int* in_sizes, int n_in,
                              void* d_out, int out_size, void* d_ws, size_t ws_size,
                              hipStream_t stream) {
    (void)in_sizes; (void)n_in; (void)out_size; (void)ws_size;
    const int NU = 100000, NM = 20000, NE = 1000000, NL = 200000;

    const float* x_user  = (const float*)d_in[0];
    const float* x_movie = (const float*)d_in[1];
    const float* P[36];                       // 4 convs x {Wq,bq,Wk,bk,Wv,bv,We,Ws,bs}
    for (int i = 0; i < 36; ++i) P[i] = (const float*)d_in[2 + i];
    const float* W1 = (const float*)d_in[38];
    const float* b1 = (const float*)d_in[39];
    const float* W2 = (const float*)d_in[40];
    const float* b2 = (const float*)d_in[41];
    const int* src_user  = (const int*)d_in[42];
    const int* dst_movie = (const int*)d_in[43];
    const float* edge_attr = (const float*)d_in[44];
    const int* label_src = (const int*)d_in[45];
    const int* label_dst = (const int*)d_in[46];

    float* ws = (float*)d_ws;
    size_t o = 0;
    float* h_u  = ws + o; o += (size_t)NU * 64;
    float* h_m  = ws + o; o += (size_t)NM * 64;
    float* z_u  = ws + o; o += (size_t)NU * 64;
    float* z_m  = ws + o; o += (size_t)NM * 64;
    float* Qb   = ws + o; o += (size_t)NU * 64;
    float* Kb   = ws + o; o += (size_t)NU * 64;
    float* Vb   = ws + o; o += (size_t)NU * 64;
    float* alog = ws + o; o += (size_t)NE;
    float* ssum = ws + o; o += (size_t)NU;
    unsigned* mmax = (unsigned*)(ws + o); o += (size_t)NU;

    auto conv = [&](const float* const* p, const float* xs, const float* xd,
                    const int* si, const int* di, int ns, int nd,
                    float* outb, bool relu) {
        const float *Wq = p[0], *bq = p[1], *Wk = p[2], *bk = p[3],
                    *Wv = p[4], *bv = p[5], *We = p[6], *Ws = p[7], *bs = p[8];
        proj64<<<(nd + 127) / 128, 256, 0, stream>>>(xd, Wq, bq, Qb, nd);
        proj64<<<(ns + 127) / 128, 256, 0, stream>>>(xs, Wk, bk, Kb, ns);
        proj64<<<(ns + 127) / 128, 256, 0, stream>>>(xs, Wv, bv, Vb, ns);
        proj64<<<(nd + 127) / 128, 256, 0, stream>>>(xd, Ws, bs, outb, nd); // skip
        init_ms<<<(nd + 255) / 256, 256, 0, stream>>>(mmax, ssum, nd);
        attn_logits<<<(NE + 7) / 8, 256, 0, stream>>>(Qb, Kb, si, di, edge_attr,
                                                      We, alog, mmax, NE);
        attn_exp<<<(NE + 255) / 256, 256, 0, stream>>>(alog, mmax, ssum, di, NE);
        attn_agg<<<(NE + 3) / 4, 256, 0, stream>>>(Vb, alog, ssum, We, edge_attr,
                                                   si, di, outb, NE);
        if (relu)
            relu_k<<<((size_t)nd * 64 + 255) / 256, 256, 0, stream>>>(outb, nd * 64);
    };

    // layer 1
    conv(P + 0,  x_user,  x_movie, src_user,  dst_movie, NU, NM, h_m, true);
    conv(P + 9,  x_movie, x_user,  dst_movie, src_user,  NM, NU, h_u, true);
    // layer 2
    conv(P + 18, h_u, h_m, src_user,  dst_movie, NU, NM, z_m, false);
    conv(P + 27, h_m, h_u, dst_movie, src_user,  NM, NU, z_u, false);
    // edge decoder
    decoder<<<(NL / 16 + 3) / 4, 128, 0, stream>>>(z_u, z_m, label_src, label_dst,
                                                   W1, b1, W2, b2, (float*)d_out, NL);
}
